// co_matrix_add_SkipGram_36223754174522
// MI455X (gfx1250) — matrix operand order repaired and checked by reading, not compiled, not run on hardware
//
#include <hip/hip_runtime.h>

typedef __attribute__((ext_vector_type(16))) __bf16 v16bf;
typedef __attribute__((ext_vector_type(8)))  __bf16 v8bf;
typedef __attribute__((ext_vector_type(8)))  float  v8f;
typedef __attribute__((ext_vector_type(4)))  float  f4;

namespace {
constexpr int V  = 8192;
constexpr int H  = 128;
constexpr int B  = 1024;
constexpr int C  = 8;
constexpr int MT = 32;             // output rows per block (smaller tile -> 288 blocks)
constexpr int KC = 64;             // K chunk staged in LDS (2 WMMA K-steps)
constexpr int LDB = KC + 8;        // bf16 LDS row stride: 144 B, b128-aligned, bank-spread
constexpr int CTX_BLOCKS = (B * C) / MT;   // 256
constexpr int TGT_BLOCKS = B / MT;         // 32
constexpr size_t HV = (size_t)H * V;       // 1M elements per W matrix
constexpr size_t WS_NEED = 4 * HV * sizeof(__bf16);  // hi/lo planes for both W's: 8 MB
}

// ---------------- pre-pass: split W into hi/lo bf16 planes in workspace --------------
// ws layout (bf16 elements): [W_in_hi | W_in_lo | W_tg_hi | W_tg_lo], each H*V.
__global__ __launch_bounds__(256)
void wconvert(const float* __restrict__ W_in, const float* __restrict__ W_tg,
              __bf16* __restrict__ ws) {
    size_t g = ((size_t)blockIdx.x * 256 + threadIdx.x) * 8;
    const float* src;
    __bf16 *dhi, *dlo;
    if (g < HV) { src = W_in + g;        dhi = ws + g;          dlo = ws + HV + g; }
    else        { size_t e = g - HV;
                  src = W_tg + e;        dhi = ws + 2 * HV + e; dlo = ws + 3 * HV + e; }
    f4 x0 = *(const f4*)(src);
    f4 x1 = *(const f4*)(src + 4);
    union { v8bf v; __bf16 e[8]; } hh, ll;
#pragma unroll
    for (int i = 0; i < 4; ++i) {
        __bf16 h0 = (__bf16)x0[i];
        hh.e[i] = h0;     ll.e[i] = (__bf16)(x0[i] - (float)h0);
        __bf16 h1 = (__bf16)x1[i];
        hh.e[4 + i] = h1; ll.e[4 + i] = (__bf16)(x1[i] - (float)h1);
    }
    *(v8bf*)(dhi) = hh.v;
    *(v8bf*)(dlo) = ll.v;
}

// fp32 -> (hi, lo) bf16 split of 8 staged values, stored to the two LDS planes
__device__ __forceinline__ void cvt_store8(const f4& a0, const f4& a1,
                                           __bf16* __restrict__ dhi, __bf16* __restrict__ dlo) {
    union { v8bf v; __bf16 e[8]; } hh, ll;
#pragma unroll
    for (int i = 0; i < 4; ++i) {
        __bf16 h0 = (__bf16)a0[i];
        hh.e[i] = h0;     ll.e[i] = (__bf16)(a0[i] - (float)h0);
        __bf16 h1 = (__bf16)a1[i];
        hh.e[4 + i] = h1; ll.e[4 + i] = (__bf16)(a1[i] - (float)h1);
    }
    *(v8bf*)(dhi) = hh.v;
    *(v8bf*)(dlo) = ll.v;
}

// fallback: per-wave fp32 -> hi/lo bf16 split for the B fragment (registers only)
__device__ __forceinline__ void split_bf16(const float* __restrict__ x /*16 vals*/,
                                           v16bf& hi, v16bf& lo) {
#pragma unroll
    for (int i = 0; i < 16; ++i) {
        float f  = x[i];
        __bf16 h = (__bf16)f;
        hi[i] = h;
        lo[i] = (__bf16)(f - (float)h);
    }
}

template <bool PRE>
__global__ __launch_bounds__(256, 2)
void skipgram_fused(const int* __restrict__ contexts, const int* __restrict__ target,
                    const float* __restrict__ co, const float* __restrict__ W_in,
                    const float* __restrict__ W_tg, const __bf16* __restrict__ wsp,
                    float* __restrict__ out)
{
    __shared__ __bf16 sHi[2][MT * LDB];   // hi-plane, double buffered (2 x 4608 B)
    __shared__ __bf16 sLo[2][MT * LDB];   // lo-plane
    __shared__ int    sIdx[MT];

    const int tid   = threadIdx.x;
    const int lane  = tid & 31;
    const int wave  = tid >> 5;           // 0..7 -> 16-wide h tile
    const int nrow  = lane & 15;
    const int khalf = lane >> 4;

    int mode, m0;
    const float* Wp;
    if ((int)blockIdx.x < CTX_BLOCKS) {
        mode = 0; m0 = blockIdx.x * MT;                     Wp = W_in;
    } else {
        mode = 1; m0 = ((int)blockIdx.x - CTX_BLOCKS) * MT; Wp = W_tg;
    }

    if (tid < MT) {
        int m = m0 + tid;
        sIdx[tid] = (mode == 0) ? contexts[m] : target[m];
    }
    __syncthreads();

    // A staging: 8 threads per gathered row, 8 f32 each, split once into bf16 planes
    const int arow = tid >> 3;                  // 0..31
    const int aseg = tid & 7;                   // 8-float segment within KC
    const size_t arow_base = (size_t)sIdx[arow] * V + (size_t)aseg * 8;
    const int    aoff      = arow * LDB + aseg * 8;

    const int h = wave * 16 + nrow;
    const float*  const wbase = Wp + (size_t)h * V;              // exact f32 row (epilogue)
    const __bf16* const whi   = PRE ? wsp + (size_t)mode * 2 * HV + (size_t)h * V : nullptr;
    const __bf16* const wlo   = PRE ? whi + HV : nullptr;

    v8f acc[2] = {};                            // 2 M-subtiles of 16x16 f32

    // ---- preload + stage chunk 0 ----
    {
        const float* src = co + arow_base;
        f4 a0 = *(const f4*)(src + 0);
        f4 a1 = *(const f4*)(src + 4);
        cvt_store8(a0, a1, &sHi[0][aoff], &sLo[0][aoff]);
    }
    __syncthreads();

    int p = 0;
    for (int k0 = 0; k0 < V; k0 += KC) {
        // ---- issue next chunk's gathered global loads early (latency under WMMAs) ----
        const bool nxt = (k0 + KC) < V;
        f4 n0, n1;
        if (nxt) {
            const float* src = co + arow_base + k0 + KC;
            n0 = *(const f4*)(src + 0);
            n1 = *(const f4*)(src + 4);
            __builtin_prefetch(src + 2 * KC, 0, 0);   // global_prefetch_b8, 2 chunks ahead
        }

        const __bf16* const bufHi = sHi[p];
        const __bf16* const bufLo = sLo[p];
#pragma unroll
        for (int kc = 0; kc < KC; kc += 32) {
            // ---- B fragment: 16 contiguous K per lane-half ----
            v16bf bhi, blo;
            if constexpr (PRE) {
                // pre-split bf16 planes, L2-resident, 32B-aligned vector loads
                bhi = *(const v16bf*)(whi + k0 + kc + khalf * 16);
                blo = *(const v16bf*)(wlo + k0 + kc + khalf * 16);
            } else {
                float bx[16];
                const float* wp = wbase + k0 + kc + khalf * 16;
                f4 b0 = *(const f4*)(wp + 0);
                f4 b1 = *(const f4*)(wp + 4);
                f4 b2 = *(const f4*)(wp + 8);
                f4 b3 = *(const f4*)(wp + 12);
#pragma unroll
                for (int i = 0; i < 4; ++i) {
                    bx[i] = b0[i]; bx[4 + i] = b1[i]; bx[8 + i] = b2[i]; bx[12 + i] = b3[i];
                }
                split_bf16(bx, bhi, blo);
            }

#pragma unroll
            for (int sub = 0; sub < 2; ++sub) {
                // A fragments: pure ds_load_b128, pre-split at staging time
                union { v16bf v; v8bf half[2]; } ahi, alo;
                {
                    const int fo = (sub * 16 + nrow) * LDB + kc + khalf * 16;
                    ahi.half[0] = *(const v8bf*)(bufHi + fo);
                    ahi.half[1] = *(const v8bf*)(bufHi + fo + 8);
                    alo.half[0] = *(const v8bf*)(bufLo + fo);
                    alo.half[1] = *(const v8bf*)(bufLo + fo + 8);
                }
                // fp32-emulated product: hi*hi + lo*hi + hi*lo (f32 accumulate)
                acc[sub] = __builtin_amdgcn_wmma_f32_16x16x32_bf16(
                    false, ahi.v, false, bhi, (short)0, acc[sub], false, false);
                acc[sub] = __builtin_amdgcn_wmma_f32_16x16x32_bf16(
                    false, alo.v, false, bhi, (short)0, acc[sub], false, false);
                acc[sub] = __builtin_amdgcn_wmma_f32_16x16x32_bf16(
                    false, ahi.v, false, blo, (short)0, acc[sub], false, false);
            }
        }

        // ---- stage next chunk into the other buffer, one barrier per iteration ----
        if (nxt) {
            cvt_store8(n0, n1, &sHi[p ^ 1][aoff], &sLo[p ^ 1][aoff]);
        }
        __syncthreads();
        p ^= 1;
    }

    // ---- epilogue: fuse one-hot add (exact f32 W[h, idx]) + transpose scatter ----
#pragma unroll
    for (int sub = 0; sub < 2; ++sub) {
#pragma unroll
        for (int r = 0; r < 8; ++r) {
            int Mi   = r + 8 * khalf;          // C/D layout: VGPR r, lane-half -> M
            int mloc = sub * 16 + Mi;
            int m    = m0 + mloc;
            float val = acc[sub][r] + wbase[sIdx[mloc]];
            size_t off;
            if (mode == 0) {                    // contexts_out is [C, B, H], m = b*C + c
                int b = m / C, c = m % C;
                off = ((size_t)(c * B + b)) * H + h;
            } else {                            // target_out appended after [C,B,H]
                off = (size_t)C * B * H + (size_t)m * H + h;
            }
            out[off] = val;
        }
    }
}

extern "C" void kernel_launch(void* const* d_in, const int* in_sizes, int n_in,
                              void* d_out, int out_size, void* d_ws, size_t ws_size,
                              hipStream_t stream) {
    (void)in_sizes; (void)n_in; (void)out_size;
    const int*   contexts = (const int*)d_in[0];
    const int*   target   = (const int*)d_in[1];
    const float* co       = (const float*)d_in[2];
    const float* W_in     = (const float*)d_in[3];
    const float* W_tg     = (const float*)d_in[4];
    float*       out      = (float*)d_out;

    dim3 grid(CTX_BLOCKS + TGT_BLOCKS);   // 288 blocks, 8 wave32 waves each

    if (d_ws != nullptr && ws_size >= WS_NEED) {
        // one-time W split: 2*H*V f32 -> hi/lo bf16 planes (1024 blocks x 2048 elements each)
        __bf16* ws = (__bf16*)d_ws;
        wconvert<<<dim3((unsigned)(2 * HV / (256 * 8))), 256, 0, stream>>>(W_in, W_tg, ws);
        skipgram_fused<true><<<grid, 256, 0, stream>>>(contexts, target, co, W_in, W_tg,
                                                       ws, out);
    } else {
        skipgram_fused<false><<<grid, 256, 0, stream>>>(contexts, target, co, W_in, W_tg,
                                                        nullptr, out);
    }
}
